// GNNAgent_31172872634844
// MI455X (gfx1250) — matrix operand order repaired and checked by reading, not compiled, not run on hardware
//
#include <hip/hip_runtime.h>
#include <hip/hip_bf16.h>

#define N_AGENTS 50
#define PAD_M    64
#define HIDDEN   128
#define N_ACT    8
#define BS       2048
#define N_PASSES 3

typedef __attribute__((ext_vector_type(16))) __bf16 v16bf;
typedef __attribute__((ext_vector_type(8)))  float  v8f;
typedef __attribute__((ext_vector_type(8)))  unsigned short u16x8;

union FragU { v16bf v; u16x8 h[2]; };

__device__ __forceinline__ unsigned short f2bf(float f) {
  unsigned int x = __float_as_uint(f);
  x += 0x7FFFu + ((x >> 16) & 1u);            // round-to-nearest-even
  return (unsigned short)(x >> 16);
}

// A-fragment (16x32 bf16), row-major source, K-contiguous.
// lanes 0-15: row M=lane, halves0-7 => K 0-7, halves8-15 => K 16-23;
// lanes 16-31: same rows, K 8-15 / 24-31.  Two 16B loads per lane.
__device__ __forceinline__ v16bf load_a_frag(const unsigned short* base, int stride,
                                             int m0, int k0, int lane) {
  const int r = lane & 15, hi = lane >> 4;
  const unsigned short* p = base + (m0 + r) * stride + k0 + 8 * hi;
  FragU u;
  u.h[0] = *(const u16x8*)(p);
  u.h[1] = *(const u16x8*)(p + 16);
  return u.v;
}

// B-fragment (32x16 bf16). Source row-major over N, K-contiguous.
// lane = column N; lanes 0-15 halves = K 0-7 / 16-23, lanes 16-31 halves = K 8-15 / 24-31.
__device__ __forceinline__ v16bf load_b_frag(const unsigned short* base, int stride,
                                             int n0, int k0, int lane) {
  const int j = lane & 15, hi = lane >> 4;
  const unsigned short* p = base + (n0 + j) * stride + k0 + 8 * hi;
  FragU u;
  u.h[0] = *(const u16x8*)(p);
  u.h[1] = *(const u16x8*)(p + 16);
  return u.v;
}

__device__ __forceinline__ v8f wmma_bf16(v16bf a, v16bf b, v8f c) {
  return __builtin_amdgcn_wmma_f32_16x16x32_bf16(false, a, false, b, (short)0, c,
                                                 false, false);
}

__device__ __forceinline__ float sigmoidf_(float x) {
  return 1.0f / (1.0f + __expf(-x));
}

// ---- prep: f32 -> bf16 with zero padding -------------------------------
__global__ void convert_pad_kernel(const float* __restrict__ src,
                                   unsigned short* __restrict__ dst,
                                   int dst_rows, int dst_cols,
                                   int src_rows, int src_cols) {
  int i = blockIdx.x * blockDim.x + threadIdx.x;
  int total = dst_rows * dst_cols;
  if (i >= total) return;
  int r = i / dst_cols, c = i - r * dst_cols;
  float v = (r < src_rows && c < src_cols) ? src[r * src_cols + c] : 0.0f;
  dst[i] = f2bf(v);
}

// dynamic-LDS layout (halves):
//   sX   [64*128]   agent-major current features
//   sY   [128*64]   feature-major (transposed) GEMM intermediate
//   sH   [64*128]   hidden (then new h) bf16
//   sAdj [64*64]    padded adjacency
//   sWg  [128*128]  staged w_gcn; later reused as GRU per-n weight stage
// then f32: sBg[128], sBih[384], sBhh[384], sBf2[8]
#define SMEM_HALVES (64*128 + 128*64 + 64*128 + 64*64 + 128*128)
#define SMEM_BYTES  (SMEM_HALVES * 2 + (128 + 384 + 384 + 8) * 4)

// ---- fused GNN-agent forward: one block per batch sample ---------------
__global__ __launch_bounds__(128) void gnn_agent_fused(
    const float* __restrict__ inputs,   // (BS*50, 128)
    const float* __restrict__ hidden,   // (BS*50, 128)
    const float* __restrict__ b_gcn,    // (128)
    const float* __restrict__ b_ih,     // (384)
    const float* __restrict__ b_hh,     // (384)
    const float* __restrict__ b_fc2,    // (8)
    const unsigned short* __restrict__ wgcn,  // bf16 (128,128)
    const unsigned short* __restrict__ wih,   // bf16 (384,128)
    const unsigned short* __restrict__ whh,   // bf16 (384,128)
    const unsigned short* __restrict__ wfc2,  // bf16 (16,128), rows 8..15 zero
    const unsigned short* __restrict__ adjbf, // bf16 (64,64), padded
    float* __restrict__ out_q,          // (BS*50, 8)
    float* __restrict__ out_h)          // (BS*50, 128)
{
  extern __shared__ __align__(16) unsigned char smem_raw[];
  unsigned short* sX   = (unsigned short*)smem_raw;
  unsigned short* sY   = sX   + PAD_M * HIDDEN;
  unsigned short* sH   = sY   + HIDDEN * PAD_M;
  unsigned short* sAdj = sH   + PAD_M * HIDDEN;
  unsigned short* sWg  = sAdj + PAD_M * PAD_M;       // 128*128 halves (32 KB)
  float* sBg  = (float*)(sWg + HIDDEN * HIDDEN);
  float* sBih = sBg  + HIDDEN;
  float* sBhh = sBih + 3 * HIDDEN;
  float* sBf2 = sBhh + 3 * HIDDEN;

  const int tid  = threadIdx.x;
  const int lane = tid & 31;
  const int w    = tid >> 5;          // wave id 0..3 -> M tile
  const int m0   = w * 16;
  const int j    = lane & 15;
  const int hi   = lane >> 4;
  const long long row0 = (long long)blockIdx.x * N_AGENTS;

  if (tid == 0) {                     // warm L2 for streamed GRU weights
    __builtin_prefetch(wih, 0, 2);
    __builtin_prefetch(whh, 0, 2);
  }

  // stage inputs / hidden (f32 -> bf16), adjacency, w_gcn, biases into LDS
  for (int i = tid; i < N_AGENTS * HIDDEN; i += 128) {
    sX[i] = f2bf(inputs[row0 * HIDDEN + i]);
    sH[i] = f2bf(hidden[row0 * HIDDEN + i]);
  }
  for (int i = N_AGENTS * HIDDEN + tid; i < PAD_M * HIDDEN; i += 128) {
    sX[i] = 0; sH[i] = 0;
  }
  for (int i = tid; i < PAD_M * PAD_M / 2; i += 128)             // dword copies
    ((unsigned int*)sAdj)[i] = ((const unsigned int*)adjbf)[i];
  for (int i = tid; i < HIDDEN * HIDDEN / 2; i += 128)
    ((unsigned int*)sWg)[i] = ((const unsigned int*)wgcn)[i];
  if (tid < HIDDEN) sBg[tid] = b_gcn[tid];
  for (int i = tid; i < 3 * HIDDEN; i += 128) { sBih[i] = b_ih[i]; sBhh[i] = b_hh[i]; }
  if (tid < N_ACT) sBf2[tid] = b_fc2[tid];

  // ---------------- 3 GCN passes: X = relu(adj @ (X @ Wg^T) + b) --------
  for (int pass = 0; pass < N_PASSES; ++pass) {
    __syncthreads();
    // Y_T = (X @ Wg^T)^T : store feature-major so next GEMM B-frags are
    // contiguous ds_load_b128s;  Wg B-frags come from LDS (staged once)
    for (int n = 0; n < 8; ++n) {
      v8f acc = {};
#pragma unroll
      for (int k = 0; k < 4; ++k) {
        v16bf a = load_a_frag(sX,  HIDDEN, m0,     k * 32, lane);
        v16bf b = load_b_frag(sWg, HIDDEN, n * 16, k * 32, lane);
        acc = wmma_bf16(a, b, acc);
      }
      u16x8 pk;
#pragma unroll
      for (int r = 0; r < 8; ++r) pk[r] = f2bf(acc[r]);
      *(u16x8*)(&sY[(n * 16 + j) * PAD_M + m0 + 8 * hi]) = pk;
    }
    __syncthreads();
    // X = relu(adj @ Y + b_gcn)
    for (int n = 0; n < 8; ++n) {
      v8f acc = {};
#pragma unroll
      for (int k = 0; k < 2; ++k) {
        v16bf a = load_a_frag(sAdj, PAD_M, m0,     k * 32, lane);
        v16bf b = load_b_frag(sY,   PAD_M, n * 16, k * 32, lane); // B[k][n]=Y_T[n][k]
        acc = wmma_bf16(a, b, acc);
      }
      float bias = sBg[n * 16 + j];
#pragma unroll
      for (int r = 0; r < 8; ++r) {
        float v = acc[r] + bias;
        v = v > 0.0f ? v : 0.0f;
        sX[(m0 + r + 8 * hi) * HIDDEN + n * 16 + j] = f2bf(v);
      }
    }
  }

  // ---------------- GRU cell -------------------------------------------
  // Per n-tile, cooperatively stage the 6 needed 16x128 weight slices
  // (wih_r/z/n, whh_r/z/n) into sWg (reused), so B-frags are shared by all
  // waves via LDS instead of being re-streamed from L2 by each wave.
  for (int n = 0; n < 8; ++n) {
    const int n0 = n * 16;
    __syncthreads();   // previous iteration's ds reads of sWg complete
    {
      const unsigned int* wihd = (const unsigned int*)wih;
      const unsigned int* whhd = (const unsigned int*)whh;
      unsigned int* dst = (unsigned int*)sWg;
      // 6 regions * 16 rows * 64 dwords
      for (int d = tid; d < 6 * 16 * 64; d += 128) {
        const int g  = d >> 10;           // region 0..5
        const int rr = (d >> 6) & 15;     // row in region
        const int c  = d & 63;            // dword in row
        const int gateRow = ((g >= 3 ? g - 3 : g) * HIDDEN) + n0 + rr;
        dst[d] = (g < 3) ? wihd[gateRow * 64 + c] : whhd[gateRow * 64 + c];
      }
    }
    __syncthreads();   // staged weights visible

    v8f acc_r = {}, acc_z = {}, acc_in = {}, acc_hn = {};
#pragma unroll
    for (int k = 0; k < 4; ++k) {
      v16bf ax  = load_a_frag(sX, HIDDEN, m0, k * 32, lane);
      v16bf ah  = load_a_frag(sH, HIDDEN, m0, k * 32, lane);
      v16bf brx = load_b_frag(sWg + 0 * 2048, HIDDEN, 0, k * 32, lane);
      v16bf bzx = load_b_frag(sWg + 1 * 2048, HIDDEN, 0, k * 32, lane);
      v16bf bnx = load_b_frag(sWg + 2 * 2048, HIDDEN, 0, k * 32, lane);
      v16bf brh = load_b_frag(sWg + 3 * 2048, HIDDEN, 0, k * 32, lane);
      v16bf bzh = load_b_frag(sWg + 4 * 2048, HIDDEN, 0, k * 32, lane);
      v16bf bnh = load_b_frag(sWg + 5 * 2048, HIDDEN, 0, k * 32, lane);
      acc_r  = wmma_bf16(ax, brx, acc_r);
      acc_r  = wmma_bf16(ah, brh, acc_r);   // gi_r + gh_r fused (K=256 chain)
      acc_z  = wmma_bf16(ax, bzx, acc_z);
      acc_z  = wmma_bf16(ah, bzh, acc_z);
      acc_in = wmma_bf16(ax, bnx, acc_in);  // i_n alone
      acc_hn = wmma_bf16(ah, bnh, acc_hn);  // h_n alone (needed for r*h_n)
    }
    const int col   = n0 + j;
    const float br  = sBih[col]          + sBhh[col];
    const float bz  = sBih[HIDDEN + col] + sBhh[HIDDEN + col];
    const float bin = sBih[2 * HIDDEN + col];
    const float bhn = sBhh[2 * HIDDEN + col];
#pragma unroll
    for (int r = 0; r < 8; ++r) {
      const int rowt = m0 + r + 8 * hi;
      const float rr = sigmoidf_(acc_r[r] + br);
      const float zz = sigmoidf_(acc_z[r] + bz);
      const float nn = tanhf(acc_in[r] + bin + rr * (acc_hn[r] + bhn));
      float hin = 0.0f;
      if (rowt < N_AGENTS) hin = hidden[(row0 + rowt) * HIDDEN + col];  // f32
      const float h = (1.0f - zz) * nn + zz * hin;
      if (rowt < N_AGENTS) out_h[(row0 + rowt) * HIDDEN + col] = h;
      sH[rowt * HIDDEN + col] = f2bf(h);    // own rows only -> no cross-wave race
    }
  }

  // ---------------- q = h @ Wfc2^T + b ---------------------------------
  v8f accq = {};
#pragma unroll
  for (int k = 0; k < 4; ++k) {
    v16bf a = load_a_frag(sH, HIDDEN, m0, k * 32, lane);
    v16bf b = load_b_frag(wfc2, HIDDEN, 0, k * 32, lane);  // tiny, L2-resident
    accq = wmma_bf16(a, b, accq);
  }
  if (j < N_ACT) {
    const float bq = sBf2[j];
#pragma unroll
    for (int r = 0; r < 8; ++r) {
      const int rowt = m0 + r + 8 * hi;
      if (rowt < N_AGENTS) out_q[(row0 + rowt) * N_ACT + j] = accq[r] + bq;
    }
  }
}

extern "C" void kernel_launch(void* const* d_in, const int* in_sizes, int n_in,
                              void* d_out, int out_size, void* d_ws, size_t ws_size,
                              hipStream_t stream) {
  const float* inputs   = (const float*)d_in[0];
  const float* hidden   = (const float*)d_in[1];
  const float* norm_adj = (const float*)d_in[2];
  const float* w_gcn    = (const float*)d_in[3];
  const float* b_gcn    = (const float*)d_in[4];
  const float* w_ih     = (const float*)d_in[5];
  const float* b_ih     = (const float*)d_in[6];
  const float* w_hh     = (const float*)d_in[7];
  const float* b_hh     = (const float*)d_in[8];
  const float* w_fc2    = (const float*)d_in[9];
  const float* b_fc2    = (const float*)d_in[10];

  float* out_q = (float*)d_out;
  float* out_h = out_q + (size_t)BS * N_AGENTS * N_ACT;

  // bf16 weight staging in workspace (~242 KB)
  unsigned short* ws   = (unsigned short*)d_ws;
  unsigned short* wgcn = ws;                         // 128*128
  unsigned short* wih  = wgcn + 128 * 128;           // 384*128
  unsigned short* whh  = wih  + 384 * 128;           // 384*128
  unsigned short* wfc2 = whh  + 384 * 128;           // 16*128 (padded)
  unsigned short* adjb = wfc2 + 16 * 128;            // 64*64  (padded)

  auto launch_cvt = [&](const float* src, unsigned short* dst,
                        int dr, int dc, int sr, int sc) {
    int total = dr * dc;
    convert_pad_kernel<<<(total + 255) / 256, 256, 0, stream>>>(src, dst, dr, dc, sr, sc);
  };
  launch_cvt(w_gcn,    wgcn, HIDDEN, HIDDEN, HIDDEN, HIDDEN);
  launch_cvt(w_ih,     wih,  3 * HIDDEN, HIDDEN, 3 * HIDDEN, HIDDEN);
  launch_cvt(w_hh,     whh,  3 * HIDDEN, HIDDEN, 3 * HIDDEN, HIDDEN);
  launch_cvt(w_fc2,    wfc2, 16, HIDDEN, N_ACT, HIDDEN);
  launch_cvt(norm_adj, adjb, PAD_M, PAD_M, N_AGENTS, N_AGENTS);

  gnn_agent_fused<<<BS, 128, SMEM_BYTES, stream>>>(inputs, hidden,
                                                   b_gcn, b_ih, b_hh, b_fc2,
                                                   wgcn, wih, whh, wfc2, adjb,
                                                   out_q, out_h);
}
